// CharLSTM_76519137345625
// MI455X (gfx1250) — compile-verified, matrix operand order checked by reading, not run on hardware
//
#include <hip/hip_runtime.h>

#define SEQ   4096
#define EMB   1024
#define HID   2048
#define NCH   256
#define KTOT  (EMB + HID)      // 3072
#define KT_G  (KTOT / 32)      // 96 K-tiles per gate GEMV
#define NT_G  (HID / 16)       // 128 N-tiles per gate
#define KT_Y  (HID / 32)       // 64 K-tiles for output GEMV
#define NT_Y  (NCH / 16)       // 16 N-tiles for output GEMV
#define NBLK  64
#define NTHR  256
#define TILE_ELEMS 512         // 32x16 bf16 tile = 1 KB

typedef __attribute__((ext_vector_type(16))) __bf16 v16bf;
typedef __attribute__((ext_vector_type(8)))  float  v8f;

union TileFrag { uint4 u[2]; unsigned short s[16]; v16bf v; };

// fp32 -> bf16 round-to-nearest-even
__device__ __forceinline__ unsigned short f2bf(float f) {
  unsigned u = __float_as_uint(f);
  u += 0x7FFFu + ((u >> 16) & 1u);
  return (unsigned short)(u >> 16);
}

// Short-latency activations using native v_rcp_f32 (recurrence critical path).
__device__ __forceinline__ float fast_sigmoid(float x) {
  return __builtin_amdgcn_rcpf(1.f + __expf(-x));
}
__device__ __forceinline__ float fast_tanh(float x) {
  float a = fabsf(x);
  float e = __expf(-2.f * a);                          // e in (0,1], no overflow
  float r = (1.f - e) * __builtin_amdgcn_rcpf(1.f + e);
  return copysignf(r, x);
}

// K index inside a 32-deep tile for (lane_half, element j), matching the
// 16-bit A/B fragment VGPR order: VGPR0-3 hold K half*8+(0..7) (2 per VGPR),
// VGPR4-7 hold K 16+half*8+(0..7).
__device__ __forceinline__ int kmap(int half, int j) {
  return (j < 8) ? (half * 8 + j) : (16 + half * 8 + (j - 8));
}

// ---- weight conversion: fp32 row-major -> bf16 swizzled tiles ----
// Tile order per gate: tile(gate, ntile, ktile) at
//   ((gate*NT_G + ntile)*KT_G + ktile)*512  -> 96 KB contiguous K-stream per N-tile.
__global__ void cvt_gates(const float* __restrict__ Wf, const float* __restrict__ Wi,
                          const float* __restrict__ Wo, const float* __restrict__ Wc,
                          unsigned short* __restrict__ dst) {
  long long idx = (long long)blockIdx.x * blockDim.x + threadIdx.x; // < 4*3072*2048
  int e      = (int)(idx & 511);
  int tile   = (int)(idx >> 9);
  int kt     = tile % KT_G;
  int rest   = tile / KT_G;
  int nt     = rest & (NT_G - 1);
  int gate   = rest >> 7;
  int lane   = e >> 4, j = e & 15;
  int k = kt * 32 + kmap(lane >> 4, j);
  int n = nt * 16 + (lane & 15);
  const float* W = (gate == 0) ? Wf : (gate == 1) ? Wi : (gate == 2) ? Wo : Wc;
  dst[idx] = f2bf(W[(long long)k * HID + n]);
}

__global__ void cvt_wy(const float* __restrict__ Wy, unsigned short* __restrict__ dst) {
  int idx  = blockIdx.x * blockDim.x + threadIdx.x;  // < 16*64*512
  int e    = idx & 511;
  int tile = idx >> 9;
  int kt   = tile & (KT_Y - 1);
  int nt   = tile >> 6;
  int lane = e >> 4, j = e & 15;
  int k = kt * 32 + kmap(lane >> 4, j);
  int n = nt * 16 + (lane & 15);
  dst[idx] = f2bf(Wy[k * NCH + n]);
}

__global__ void cvt_emb(const float* __restrict__ emb, unsigned short* __restrict__ dst) {
  int i = blockIdx.x * blockDim.x + threadIdx.x;     // < 256*1024
  dst[i] = f2bf(emb[i]);
}

__global__ void init_state(const float* __restrict__ h0, const float* __restrict__ c0,
                           unsigned short* __restrict__ hb, float* __restrict__ hf,
                           float* __restrict__ cs, unsigned* __restrict__ sync) {
  int i = blockIdx.x * blockDim.x + threadIdx.x;
  if (i < HID) {
    hf[i] = h0[i];
    cs[i] = c0[i];
    hb[i]       = f2bf(h0[i]);
    hb[HID + i] = 0;
  }
  if (i == 0) { sync[0] = 0; sync[1] = 0; }
}

// ---- persistent LSTM kernel: one device-wide barrier per timestep ----
__global__ void __launch_bounds__(NTHR)
lstm_persistent(const int* __restrict__ seq,
                const unsigned short* __restrict__ Wg,
                const unsigned short* __restrict__ WyT,
                const unsigned short* __restrict__ embB,
                unsigned short* __restrict__ hb,        // [2][HID] bf16 double buffer
                float* __restrict__ hf,
                float* __restrict__ cs,
                unsigned* __restrict__ sync,            // [0]=counter [1]=generation
                const float* __restrict__ bf_, const float* __restrict__ bi_,
                const float* __restrict__ bo_, const float* __restrict__ bc_,
                const float* __restrict__ by_,
                float* __restrict__ out) {
  __shared__ unsigned short z_lds[KTOT];   // z = [x_t, h_t] as bf16
  __shared__ float g_lds[4][32];           // gate activations for this block's 32 units
  __shared__ float y_lds[8][16];           // per-wave partial y sums

  const int tid  = threadIdx.x;
  const int blk  = blockIdx.x;
  const int lane = tid & 31;
  const int wv   = tid >> 5;
  const int half = lane >> 4;

  const int gate = wv & 3;                 // 0=f 1=i 2=o 3=c
  const int ntl  = wv >> 2;                // 0..1
  const int gnt  = blk * 2 + ntl;          // global N-tile within this gate
  const float* bptr = (gate == 0) ? bf_ : (gate == 1) ? bi_ : (gate == 2) ? bo_ : bc_;
  const unsigned short* gbase = Wg + ((size_t)(gate * NT_G + gnt) * KT_G) * TILE_ELEMS;

  unsigned my_gen = 0;

  #pragma unroll 1
  for (int t = 0; t < SEQ; ++t) {
    const int cur = t & 1, nxt = cur ^ 1;
    const int row = seq[t];
    const unsigned short* embRow = embB + (size_t)row * EMB;
    const unsigned short* hcur   = hb + cur * HID;

    // stage z into LDS
    for (int k = tid; k < KTOT; k += NTHR)
      z_lds[k] = (k < EMB) ? embRow[k] : hcur[k - EMB];
    __syncthreads();

    // ---- gate GEMV via WMMA ----
    // Every lane loads the same z chunk, so ALL 16 A-rows equal z; row 0 of D
    // is the exact GEMV result (rows are independent dot products). A-fragment
    // is two contiguous ds_load_b128, B-tile is two global_load_b128.
    v8f acc = {};
    #pragma unroll 4
    for (int kt = 0; kt < KT_G; ++kt) {
      const unsigned short* tp = gbase + (size_t)kt * TILE_ELEMS;
      __builtin_prefetch(tp + 2 * TILE_ELEMS, 0, 1);   // global_prefetch ahead
      TileFrag a, b;
      const uint4* t4 = (const uint4*)tp;
      b.u[0] = t4[lane * 2 + 0];
      b.u[1] = t4[lane * 2 + 1];
      const unsigned short* zp = z_lds + kt * 32 + half * 8; // 16-B aligned
      a.u[0] = *(const uint4*)(zp);        // K = kt*32 + half*8 + (0..7)
      a.u[1] = *(const uint4*)(zp + 16);   // K = kt*32 + 16 + half*8 + (0..7)
      acc = __builtin_amdgcn_wmma_f32_16x16x32_bf16(false, a.v, false, b.v,
                                                    (short)0, acc, false, false);
    }

    // D row M=0 lives in acc[0] on lanes 0..15 (N = lane)
    if (lane < 16) {
      int n = blk * 32 + ntl * 16 + lane;
      float pre = acc[0] + bptr[n];
      float act = (gate == 3) ? fast_tanh(pre) : fast_sigmoid(pre);
      g_lds[gate][ntl * 16 + lane] = act;
    }
    __syncthreads();

    // ---- combine gates, update this block's 32 units of c/h ----
    if (tid < 32) {
      int n = blk * 32 + tid;
      float f  = g_lds[0][tid], ig = g_lds[1][tid];
      float o  = g_lds[2][tid], g  = g_lds[3][tid];
      float cn = f * cs[n] + ig * g;
      float hn = o * fast_tanh(cn);
      cs[n] = cn;
      hf[n] = hn;
      hb[nxt * HID + n] = f2bf(hn);
    }
    __threadfence();
    __syncthreads();

    // ---- device-wide barrier: h(t+1) visible to all WGPs ----
    if (tid == 0) {
      ++my_gen;
      unsigned arrived = atomicAdd(&sync[0], 1u);
      if (arrived == NBLK - 1) {
        sync[0] = 0;
        __threadfence();
        atomicAdd(&sync[1], 1u);
      } else {
        while (((volatile unsigned*)sync)[1] < my_gen) __builtin_amdgcn_s_sleep(1);
      }
      __threadfence();
    }
    __syncthreads();

    // ---- y_t = h(t+1) @ Wy + by : blocks 0..15, 8 waves split K ----
    if (blk < NT_Y) {
      const unsigned short* hn = hb + nxt * HID;       // 16-B aligned
      const unsigned short* ybase = WyT + (size_t)blk * KT_Y * TILE_ELEMS;
      v8f yac = {};
      #pragma unroll 4
      for (int kt = wv * (KT_Y / 8); kt < (wv + 1) * (KT_Y / 8); ++kt) {
        const unsigned short* tp = ybase + (size_t)kt * TILE_ELEMS;
        TileFrag a, b;
        const uint4* t4 = (const uint4*)tp;
        b.u[0] = t4[lane * 2 + 0];
        b.u[1] = t4[lane * 2 + 1];
        const unsigned short* hp = hn + kt * 32 + half * 8;
        a.u[0] = *(const uint4*)(hp);
        a.u[1] = *(const uint4*)(hp + 16);
        yac = __builtin_amdgcn_wmma_f32_16x16x32_bf16(false, a.v, false, b.v,
                                                      (short)0, yac, false, false);
      }
      if (lane < 16) y_lds[wv][lane] = yac[0];
      __syncthreads();
      if (tid < 16) {
        float s = by_[blk * 16 + tid];
        #pragma unroll
        for (int w = 0; w < 8; ++w) s += y_lds[w][tid];
        out[(size_t)t * NCH + blk * 16 + tid] = s;
      }
    }
  }

  // final h, c (each block owns the 32 units it produced)
  if (tid < 32) {
    int n = blk * 32 + tid;
    out[(size_t)SEQ * NCH + n]       = hf[n];
    out[(size_t)SEQ * NCH + HID + n] = cs[n];
  }
}

extern "C" void kernel_launch(void* const* d_in, const int* in_sizes, int n_in,
                              void* d_out, int out_size, void* d_ws, size_t ws_size,
                              hipStream_t stream) {
  const int*   seq = (const int*)  d_in[0];
  const float* h0  = (const float*)d_in[1];
  const float* c0  = (const float*)d_in[2];
  const float* emb = (const float*)d_in[3];
  const float* Wf  = (const float*)d_in[4];
  const float* bf_ = (const float*)d_in[5];
  const float* Wi  = (const float*)d_in[6];
  const float* bi_ = (const float*)d_in[7];
  const float* Wo  = (const float*)d_in[8];
  const float* bo_ = (const float*)d_in[9];
  const float* Wc  = (const float*)d_in[10];
  const float* bc_ = (const float*)d_in[11];
  const float* Wy  = (const float*)d_in[12];
  const float* by_ = (const float*)d_in[13];
  float* out = (float*)d_out;

  // workspace layout (bytes): ~52 MB total, weights live in bf16 (fits in 192 MB L2)
  char* ws = (char*)d_ws;
  unsigned short* Wg   = (unsigned short*)(ws);                 // 4*3072*2048*2 = 50331648
  unsigned short* WyT  = (unsigned short*)(ws + 50331648);      // 2048*256*2    = 1048576
  unsigned short* embB = (unsigned short*)(ws + 51380224);      // 256*1024*2    = 524288
  unsigned short* hbuf = (unsigned short*)(ws + 51904512);      // 2*2048*2      = 8192
  float*          hf   = (float*)        (ws + 51912704);      // 2048*4        = 8192
  float*          cst  = (float*)        (ws + 51920896);      // 2048*4        = 8192
  unsigned*       sync = (unsigned*)     (ws + 51929088);      // 2*4

  cvt_gates <<<98304, 256, 0, stream>>>(Wf, Wi, Wo, Wc, Wg);
  cvt_wy    <<<2048,  256, 0, stream>>>(Wy, WyT);
  cvt_emb   <<<1024,  256, 0, stream>>>(emb, embB);
  init_state<<<8,     256, 0, stream>>>(h0, c0, hbuf, hf, cst, sync);
  lstm_persistent<<<NBLK, NTHR, 0, stream>>>(seq, Wg, WyT, embB, hbuf, hf, cst, sync,
                                             bf_, bi_, bo_, bc_, by_, out);
  (void)in_sizes; (void)n_in; (void)out_size; (void)ws_size;
}
